// BiLSTM_CRF_8160437863180
// MI455X (gfx1250) — matrix operand order repaired and checked by reading, not compiled, not run on hardware
//
#include <hip/hip_runtime.h>
#include <hip/hip_bf16.h>

// ---------------------------------------------------------------------------
// BiLSTM-CRF on MI455X (gfx1250, wave32).
//   T=512, E=1024, HD=512 (4HD=2048 gates/dir), K=32 tags, START=30, STOP=31
// Pipeline (all on `stream`, serialized by stream order):
//   1) prep    : embedding gather + fp32->bf16 convert of x and W_ih (both dirs)
//   2) xgemm   : Xg[dir] = x @ W_ih[dir]^T + (b_ih+b_hh)   via v_wmma bf16
//   3) lstm    : 2 persistent workgroups (fwd/bwd), 512 sequential steps,
//                h/gates in LDS, c in registers, W_hh fp32 streamed from L2
//   4) tag     : feats = [hf|hb] @ w_tag^T + b_tag
//   5) viterbi : single wave32, one tag per lane, backtrack on lane 0
// ---------------------------------------------------------------------------

#define T_LEN 512
#define E_DIM 1024
#define HD    512
#define G4    2048   // 4*HD
#define K_TAG 32
#define TAG_START 30
#define TAG_STOP  31
#define NEG_INF  -10000.0f

typedef __attribute__((ext_vector_type(16))) __bf16 v16bf;
typedef __attribute__((ext_vector_type(8)))  __bf16 v8bf;
typedef __attribute__((ext_vector_type(8)))  float  v8f;

// round-to-nearest-even fp32 -> bf16 (bit exact, no __bf16 arithmetic needed)
static __device__ __forceinline__ unsigned short f2bf(float f) {
  unsigned int u = __float_as_uint(f);
  u += 0x7FFFu + ((u >> 16) & 1u);
  return (unsigned short)(u >> 16);
}

// --------------------------- 1) prep ---------------------------------------
// x_bf[t][e] = bf16(embed[sentence[t]][e]); wih_{f,b} -> bf16
__global__ void prep_kernel(const int* __restrict__ sent,
                            const float* __restrict__ embed,
                            const float* __restrict__ wih_f,
                            const float* __restrict__ wih_b,
                            unsigned short* __restrict__ x_bf,
                            unsigned short* __restrict__ wf_bf,
                            unsigned short* __restrict__ wb_bf) {
  const long long n_x = (long long)T_LEN * E_DIM;        // 524288
  const long long n_w = (long long)G4 * E_DIM;           // 2097152
  const long long total = n_x + 2 * n_w;
  long long stride = (long long)gridDim.x * blockDim.x;
  for (long long i = (long long)blockIdx.x * blockDim.x + threadIdx.x;
       i < total; i += stride) {
    if (i < n_x) {
      int t = (int)(i >> 10);
      int e = (int)(i & (E_DIM - 1));
      x_bf[i] = f2bf(embed[(long long)sent[t] * E_DIM + e]);
    } else if (i < n_x + n_w) {
      long long j = i - n_x;
      wf_bf[j] = f2bf(wih_f[j]);
    } else {
      long long j = i - n_x - n_w;
      wb_bf[j] = f2bf(wih_b[j]);
    }
  }
}

// --------------------------- 2) WMMA input GEMM ----------------------------
// Y[t][j] = sum_k x[t][k]*W[j][k] + b_ih[j] + b_hh[j]
// M=512, N=2048, K=1024, both directions. One 16x16 tile per wave,
// k-loop of 32 x v_wmma_f32_16x16x32_bf16.
__global__ __launch_bounds__(256) void
xgemm_kernel(const unsigned short* __restrict__ x_bf,
             const unsigned short* __restrict__ wf_bf,
             const unsigned short* __restrict__ wb_bf,
             const float* __restrict__ bih_f, const float* __restrict__ bhh_f,
             const float* __restrict__ bih_b, const float* __restrict__ bhh_b,
             float* __restrict__ xg_f, float* __restrict__ xg_b) {
  const int wave = threadIdx.x >> 5;          // 8 waves / block
  const int lane = threadIdx.x & 31;
  const int r    = lane & 15;
  const int half = lane >> 4;

  int gtile = blockIdx.x * 8 + wave;          // 0..8191
  int dir   = gtile >> 12;                    // 4096 tiles per direction
  int tile  = gtile & 4095;
  int mt    = tile >> 7;                      // 0..31  (T/16)
  int nt    = tile & 127;                     // 0..127 (2048/16)

  const __bf16* X = (const __bf16*)x_bf;
  const __bf16* W = (const __bf16*)(dir ? wb_bf : wf_bf);
  const float*  bi = dir ? bih_b : bih_f;
  const float*  bh = dir ? bhh_b : bhh_f;
  float*        Y  = dir ? xg_b : xg_f;

  const int m0 = mt * 16, n0 = nt * 16;
  const __bf16* arow = X + (long long)(m0 + r) * E_DIM;   // A row (M=r)
  const __bf16* brow = W + (long long)(n0 + r) * E_DIM;   // B col (N=r)

  v8f acc = {};
  for (int k0 = 0; k0 < E_DIM; k0 += 32) {
    // A 16x32 bf16: lane half picks contiguous 16-element K chunk
    v16bf A = *(const v16bf*)(arow + k0 + half * 16);
    // B 32x16 bf16: lane half picks contiguous 16-element K chunk
    v16bf B = *(const v16bf*)(brow + k0 + half * 16);
    acc = __builtin_amdgcn_wmma_f32_16x16x32_bf16(
        /*neg_a=*/false, A, /*neg_b=*/false, B,
        /*c_mod=*/(short)0, acc, /*reuse_a=*/false, /*reuse_b=*/false);
  }
  // C/D layout: lane r holds column n0+r; half selects rows 0..7 / 8..15
  const float bias = bi[n0 + r] + bh[n0 + r];
  #pragma unroll
  for (int i = 0; i < 8; ++i)
    Y[(long long)(m0 + half * 8 + i) * G4 + (n0 + r)] = acc[i] + bias;
}

// --------------------------- 3) recurrence ---------------------------------
// blockIdx.x = direction (0=fwd, 1=bwd). 1024 threads: thread owns gate
// rows j and j+1024 (fp32 GEMV vs LDS-resident h), then threads<512 do the
// cell update; c-state lives in a register of its owning thread.
__global__ __launch_bounds__(1024) void
lstm_kernel(const float* __restrict__ whh_f, const float* __restrict__ whh_b,
            const float* __restrict__ h0, const float* __restrict__ c0,
            const float* __restrict__ xg_f, const float* __restrict__ xg_b,
            float* __restrict__ hs_f, float* __restrict__ hs_b) {
  const int dir = blockIdx.x;
  const float* Whh = dir ? whh_b : whh_f;        // [2048, 512] fp32 (L2-resident)
  const float* Xg  = dir ? xg_b  : xg_f;         // [T, 2048]
  float*       hs  = dir ? hs_b  : hs_f;         // [T, 512]

  __shared__ float h_sh[HD];
  __shared__ float g_sh[G4];

  const int tid = threadIdx.x;
  if (tid < HD) h_sh[tid] = h0[dir * HD + tid];
  float c_state = (tid < HD) ? c0[dir * HD + tid] : 0.0f;
  __syncthreads();

  const int j0 = tid;                 // gates 0..1023   (i | f blocks)
  const int j1 = tid + 1024;         // gates 1024..2047 (g | o blocks)
  const float* w0 = Whh + (long long)j0 * HD;
  const float* w1 = Whh + (long long)j1 * HD;

  for (int s = 0; s < T_LEN; ++s) {
    const int t = dir ? (T_LEN - 1 - s) : s;
    float d0 = 0.0f, d1 = 0.0f;
    #pragma unroll 8
    for (int k = 0; k < HD; ++k) {
      const float hk = h_sh[k];       // LDS broadcast, no bank conflicts
      d0 = fmaf(w0[k], hk, d0);
      d1 = fmaf(w1[k], hk, d1);
    }
    g_sh[j0] = Xg[(long long)t * G4 + j0] + d0;
    g_sh[j1] = Xg[(long long)t * G4 + j1] + d1;
    __syncthreads();
    if (tid < HD) {
      const float ig = 1.0f / (1.0f + __expf(-g_sh[tid]));
      const float fg = 1.0f / (1.0f + __expf(-g_sh[HD + tid]));
      const float gg = tanhf(g_sh[2 * HD + tid]);
      const float og = 1.0f / (1.0f + __expf(-g_sh[3 * HD + tid]));
      c_state = fg * c_state + ig * gg;
      const float hn = og * tanhf(c_state);
      h_sh[tid] = hn;
      hs[(long long)t * HD + tid] = hn;
    }
    __syncthreads();
  }
}

// --------------------------- 4) tag projection -----------------------------
__global__ __launch_bounds__(32) void
tag_kernel(const float* __restrict__ hs_f, const float* __restrict__ hs_b,
           const float* __restrict__ w_tag, const float* __restrict__ b_tag,
           float* __restrict__ feats) {
  const int t = blockIdx.x;
  const int k = threadIdx.x;                 // one wave32, one tag per lane
  const float* wr = w_tag + (long long)k * (2 * HD);
  const float* hf = hs_f + (long long)t * HD;
  const float* hb = hs_b + (long long)t * HD;
  float acc = b_tag[k];
  #pragma unroll 8
  for (int i = 0; i < HD; ++i) acc = fmaf(hf[i], wr[i], acc);
  #pragma unroll 8
  for (int i = 0; i < HD; ++i) acc = fmaf(hb[i], wr[HD + i], acc);
  feats[t * K_TAG + k] = acc;
}

// --------------------------- 5) Viterbi ------------------------------------
// One wave32; lane j owns tag j. First-occurrence argmax (strict >) to match
// jnp.argmax tie behavior.
__global__ __launch_bounds__(32) void
viterbi_kernel(const float* __restrict__ feats,
               const float* __restrict__ trans,
               int* __restrict__ bp, float* __restrict__ out) {
  __shared__ float fv[K_TAG];
  __shared__ float nfv[K_TAG];
  __shared__ float term[K_TAG];
  const int j = threadIdx.x;

  fv[j] = (j == TAG_START) ? 0.0f : NEG_INF;
  __syncthreads();

  for (int t = 0; t < T_LEN; ++t) {
    float best = -3.4e38f;
    int bpi = 0;
    #pragma unroll
    for (int p = 0; p < K_TAG; ++p) {
      const float s = fv[p] + trans[j * K_TAG + p];   // [next, prev]
      if (s > best) { best = s; bpi = p; }
    }
    bp[t * K_TAG + j] = bpi;
    nfv[j] = best + feats[t * K_TAG + j];
    __syncthreads();
    fv[j] = nfv[j];
    __syncthreads();
  }

  term[j] = fv[j] + trans[TAG_STOP * K_TAG + j];
  __syncthreads();

  if (j == 0) {
    int bestk = 0;
    float bs = term[0];
    for (int p = 1; p < K_TAG; ++p)
      if (term[p] > bs) { bs = term[p]; bestk = p; }
    out[0] = bs;                                   // score
    out[1 + (T_LEN - 1)] = (float)bestk;           // path[T-1] = best
    int tag = bestk;
    for (int t = T_LEN - 1; t >= 1; --t) {         // path[t-1] = bp[t][path[t]]
      tag = bp[t * K_TAG + tag];
      out[1 + t - 1] = (float)tag;
    }
  }
}

// --------------------------- launch ----------------------------------------
extern "C" void kernel_launch(void* const* d_in, const int* in_sizes, int n_in,
                              void* d_out, int out_size, void* d_ws, size_t ws_size,
                              hipStream_t stream) {
  (void)in_sizes; (void)n_in; (void)out_size; (void)ws_size;

  const int*   sentence = (const int*)  d_in[0];
  const float* embed    = (const float*)d_in[1];
  const float* w_ih_f   = (const float*)d_in[2];
  const float* w_hh_f   = (const float*)d_in[3];
  const float* b_ih_f   = (const float*)d_in[4];
  const float* b_hh_f   = (const float*)d_in[5];
  const float* w_ih_b   = (const float*)d_in[6];
  const float* w_hh_b   = (const float*)d_in[7];
  const float* b_ih_b   = (const float*)d_in[8];
  const float* b_hh_b   = (const float*)d_in[9];
  const float* h0       = (const float*)d_in[10];
  const float* c0       = (const float*)d_in[11];
  const float* w_tag    = (const float*)d_in[12];
  const float* b_tag    = (const float*)d_in[13];
  const float* trans    = (const float*)d_in[14];
  float* out = (float*)d_out;

  // workspace layout (bytes)
  char* ws = (char*)d_ws;
  unsigned short* x_bf  = (unsigned short*)(ws + 0);                  // 1 MB
  unsigned short* wf_bf = (unsigned short*)(ws + (1u << 20));         // 4 MB
  unsigned short* wb_bf = (unsigned short*)(ws + 5u * (1u << 20));    // 4 MB
  float* xg_f  = (float*)(ws + 9u  * (1u << 20));                     // 4 MB
  float* xg_b  = (float*)(ws + 13u * (1u << 20));                     // 4 MB
  float* hs_f  = (float*)(ws + 17u * (1u << 20));                     // 1 MB
  float* hs_b  = (float*)(ws + 18u * (1u << 20));                     // 1 MB
  float* feats = (float*)(ws + 19u * (1u << 20));                     // 64 KB
  int*   bp    = (int*)  (ws + 19u * (1u << 20) + (1u << 16));        // 64 KB

  prep_kernel<<<4608, 256, 0, stream>>>(sentence, embed, w_ih_f, w_ih_b,
                                        x_bf, wf_bf, wb_bf);
  xgemm_kernel<<<1024, 256, 0, stream>>>(x_bf, wf_bf, wb_bf,
                                         b_ih_f, b_hh_f, b_ih_b, b_hh_b,
                                         xg_f, xg_b);
  lstm_kernel<<<2, 1024, 0, stream>>>(w_hh_f, w_hh_b, h0, c0,
                                      xg_f, xg_b, hs_f, hs_b);
  tag_kernel<<<T_LEN, 32, 0, stream>>>(hs_f, hs_b, w_tag, b_tag, feats);
  viterbi_kernel<<<1, 32, 0, stream>>>(feats, trans, bp, out);
}
